// Transformer_88811333747146
// MI455X (gfx1250) — matrix operand order repaired and checked by reading, not compiled, not run on hardware
//
#include <hip/hip_runtime.h>

// ---------------------------------------------------------------------------
// Transformer encoder-decoder forward for MI455X (gfx1250), wave32 + WMMA.
// All GEMMs: v_wmma_f32_16x16x32_f16, f16 staged in LDS (double-buffered,
// software-pipelined), f32 accumulate, compile-time fused epilogues.
// Two tile configs: BN=128 (2x4 waves, 64x32 per wave) for N%128==0 GEMMs,
// BN=64 (4x2 waves, 32x32 per wave) for the N=64 per-head GEMMs.
// ---------------------------------------------------------------------------

typedef __attribute__((ext_vector_type(16))) _Float16 v16h;
typedef __attribute__((ext_vector_type(8)))  float    v8f;

static constexpr int B_  = 8;
static constexpr int S_  = 512;
static constexpr int D_  = 512;
static constexpr int H_  = 8;
static constexpr int DH_ = 64;
static constexpr int F_  = 2048;
static constexpr int V_  = 32000;
static constexpr int NE_ = 4;
static constexpr int ND_ = 4;

static constexpr int BM   = 128;  // M tile per block (fixed)
static constexpr int KT   = 32;   // K step per WMMA
static constexpr int LDSW = 40;   // halves per LDS tile row (32 + 8 pad; rows stay 16B aligned)

enum { EPI_NONE = 0, EPI_BIAS = 1, EPI_BIAS_RELU = 2, EPI_BIAS_RESID = 3, EPI_SCORE = 4 };

// ---------------------------------------------------------------------------
// Batched WMMA GEMM, z = zh*zdiv + zl batch-offset decomposition.
// TRANS_B: B stored row-major [N,K] (compute A @ B^T).
// EPI selects a compile-time epilogue (no per-element branching).
// ---------------------------------------------------------------------------
struct GemmP {
  const float* A; const float* Bm; float* C;
  const float* bias; const float* resid; const float* mask;
  long long sA_hi, sA_lo, sB_hi, sB_lo, sC_hi, sC_lo, sBias_hi, sBias_lo, sMask_lo;
  int lda, ldb, ldc;
  int K;
  int zdiv;
  float scale;
  int causal;
};

template <int BN, int WAVES_M, int WAVES_N, bool TRANS_B, int EPI>
__global__ __launch_bounds__(256) void gemm_wmma_kernel(GemmP p) {
  constexpr int WT_M = BM / WAVES_M;       // rows per wave
  constexpr int WT_N = BN / WAVES_N;       // cols per wave
  constexpr int TI   = WT_M / 16;          // 16x16 WMMA tiles per wave (M)
  constexpr int TJ   = WT_N / 16;          // 16x16 WMMA tiles per wave (N)
  constexpr int NA   = (BM * KT / 4) / 256;  // float4 A loads per thread (=4)
  constexpr int NB   = (BN * KT / 4) / 256;  // float4 B loads per thread (BN/32)
  constexpr int CB4  = BN / 4;             // float4 columns in a non-trans B row

  __shared__ _Float16 As[2][BM * LDSW];    // As[buf][m][k]
  __shared__ _Float16 Bs[2][BN * LDSW];    // Bs[buf][n][k] (pre-transposed)

  const int z  = blockIdx.z;
  const int zh = z / p.zdiv;
  const int zl = z - zh * p.zdiv;
  const float* A    = p.A  + zh * p.sA_hi + zl * p.sA_lo;
  const float* Bm   = p.Bm + zh * p.sB_hi + zl * p.sB_lo;
  float*       C    = p.C  + zh * p.sC_hi + zl * p.sC_lo;
  const float* bias = p.bias + zh * p.sBias_hi + zl * p.sBias_lo;
  const float* mask = p.mask ? p.mask + zl * p.sMask_lo : nullptr;
  const float* res  = p.resid ? p.resid + zh * p.sC_hi + zl * p.sC_lo : nullptr;

  const int m0   = blockIdx.y * BM;
  const int n0   = blockIdx.x * BN;
  const int tid  = threadIdx.x;
  const int lane = tid & 31;
  const int wave = tid >> 5;
  const int wm   = wave / WAVES_N;
  const int wn   = wave % WAVES_N;
  const int hsel = lane >> 4;
  const int l16  = lane & 15;

  // register staging for the software pipeline
  float4 ar[NA], br[NB];

  auto loadA = [&](int k0) {
#pragma unroll
    for (int i = 0; i < NA; ++i) {
      const int idx = tid + i * 256;           // BM rows x 8 float4
      const int r = idx >> 3, c4 = idx & 7;
      ar[i] = *(const float4*)(A + (long long)(m0 + r) * p.lda + k0 + c4 * 4);
    }
  };
  auto loadB = [&](int k0) {
    if constexpr (TRANS_B) {
#pragma unroll
      for (int i = 0; i < NB; ++i) {           // BN rows x 8 float4
        const int idx = tid + i * 256;
        const int r = idx >> 3, c4 = idx & 7;
        br[i] = *(const float4*)(Bm + (long long)(n0 + r) * p.ldb + k0 + c4 * 4);
      }
    } else {
#pragma unroll
      for (int i = 0; i < NB; ++i) {           // KT k-rows x CB4 float4
        const int idx = tid + i * 256;
        const int r = idx / CB4, c4 = idx % CB4;
        br[i] = *(const float4*)(Bm + (long long)(k0 + r) * p.ldb + n0 + c4 * 4);
      }
    }
  };
  auto stageA = [&](int buf) {
#pragma unroll
    for (int i = 0; i < NA; ++i) {
      const int idx = tid + i * 256;
      const int r = idx >> 3, c4 = idx & 7;
      _Float16* dst = &As[buf][r * LDSW + c4 * 4];
      dst[0] = (_Float16)ar[i].x; dst[1] = (_Float16)ar[i].y;
      dst[2] = (_Float16)ar[i].z; dst[3] = (_Float16)ar[i].w;
    }
  };
  auto stageB = [&](int buf) {
    if constexpr (TRANS_B) {
#pragma unroll
      for (int i = 0; i < NB; ++i) {
        const int idx = tid + i * 256;
        const int r = idx >> 3, c4 = idx & 7;
        _Float16* dst = &Bs[buf][r * LDSW + c4 * 4];
        dst[0] = (_Float16)br[i].x; dst[1] = (_Float16)br[i].y;
        dst[2] = (_Float16)br[i].z; dst[3] = (_Float16)br[i].w;
      }
    } else {
#pragma unroll
      for (int i = 0; i < NB; ++i) {
        const int idx = tid + i * 256;
        const int r = idx / CB4, c4 = idx % CB4;
        Bs[buf][(c4 * 4 + 0) * LDSW + r] = (_Float16)br[i].x;
        Bs[buf][(c4 * 4 + 1) * LDSW + r] = (_Float16)br[i].y;
        Bs[buf][(c4 * 4 + 2) * LDSW + r] = (_Float16)br[i].z;
        Bs[buf][(c4 * 4 + 3) * LDSW + r] = (_Float16)br[i].w;
      }
    }
  };

  v8f acc[TI][TJ] = {};

  loadA(0); loadB(0);
  stageA(0); stageB(0);
  __syncthreads();

  int buf = 0;
  for (int k0 = 0; k0 < p.K; k0 += KT) {
    const bool next = (k0 + KT) < p.K;
    if (next) { loadA(k0 + KT); loadB(k0 + KT); }   // issue globals early

    // fragments: contiguous k order (k = hsel*16 + e), same for A and B
    v16h af[TI], bf[TJ];
#pragma unroll
    for (int i = 0; i < TI; ++i) {
      const _Float16* row = &As[buf][(wm * WT_M + i * 16 + l16) * LDSW];
      // A element e -> K = hsel*16 + e
#pragma unroll
      for (int e = 0; e < 16; ++e) af[i][e] = row[hsel * 16 + e];
    }
#pragma unroll
    for (int j = 0; j < TJ; ++j) {
      const _Float16* row = &Bs[buf][(wn * WT_N + j * 16 + l16) * LDSW];
      // B element e -> K = hsel*16 + e (column = lane&15)
#pragma unroll
      for (int e = 0; e < 16; ++e) bf[j][e] = row[hsel * 16 + e];
    }
#pragma unroll
    for (int i = 0; i < TI; ++i)
#pragma unroll
      for (int j = 0; j < TJ; ++j)
        acc[i][j] = __builtin_amdgcn_wmma_f32_16x16x32_f16(
            false, af[i], false, bf[j], (short)0, acc[i][j], false, false);

    if (next) { stageA(buf ^ 1); stageB(buf ^ 1); } // wait loads + ds_store after compute
    __syncthreads();
    buf ^= 1;
  }

  // ---- compile-time epilogue ----
#pragma unroll
  for (int j = 0; j < TJ; ++j) {
    const int nn = n0 + wn * WT_N + j * 16 + l16;
    float bv = 0.0f;
    if constexpr (EPI == EPI_BIAS || EPI == EPI_BIAS_RELU || EPI == EPI_BIAS_RESID)
      bv = bias[nn];
    bool mok = true;
    if constexpr (EPI == EPI_SCORE) { if (mask) mok = mask[nn] > 0.0f; }
#pragma unroll
    for (int i = 0; i < TI; ++i) {
#pragma unroll
      for (int r = 0; r < 8; ++r) {
        const int mm = m0 + wm * WT_M + i * 16 + hsel * 8 + r;
        float v = acc[i][j][r];
        if constexpr (EPI == EPI_SCORE) {
          v *= p.scale;
          if ((p.causal && nn > mm) || !mok) v = -1.0e30f;
        } else if constexpr (EPI == EPI_BIAS) {
          v += bv;
        } else if constexpr (EPI == EPI_BIAS_RELU) {
          v = fmaxf(v + bv, 0.0f);
        } else if constexpr (EPI == EPI_BIAS_RESID) {
          v += bv + res[(long long)mm * p.ldc + nn];
        }
        C[(long long)mm * p.ldc + nn] = v;
      }
    }
  }
}

// ---------------------------------------------------------------------------
// LayerNorm over last dim (one block per row)
// ---------------------------------------------------------------------------
__global__ __launch_bounds__(128) void layernorm_kernel(
    const float* __restrict__ x, const float* __restrict__ g,
    const float* __restrict__ b, float* __restrict__ y, int D) {
  __shared__ float red[128];
  const long long row = blockIdx.x;
  const float* xr = x + row * D;
  float* yr = y + row * D;
  const int tid = threadIdx.x;

  float s = 0.f, s2 = 0.f;
  for (int d = tid; d < D; d += 128) { const float v = xr[d]; s += v; s2 += v * v; }
  red[tid] = s; __syncthreads();
  for (int off = 64; off > 0; off >>= 1) { if (tid < off) red[tid] += red[tid + off]; __syncthreads(); }
  const float mean = red[0] / D; __syncthreads();
  red[tid] = s2; __syncthreads();
  for (int off = 64; off > 0; off >>= 1) { if (tid < off) red[tid] += red[tid + off]; __syncthreads(); }
  const float var = red[0] / D - mean * mean;
  const float inv = rsqrtf(var + 1e-5f);
  for (int d = tid; d < D; d += 128) yr[d] = (xr[d] - mean) * inv * g[d] + b[d];
}

// ---------------------------------------------------------------------------
// Row softmax (in place); one block per row
// ---------------------------------------------------------------------------
__global__ __launch_bounds__(128) void softmax_kernel(float* __restrict__ sbuf, int L) {
  __shared__ float red[128];
  const long long row = blockIdx.x;
  float* r = sbuf + row * (long long)L;
  const int tid = threadIdx.x;

  float mx = -1.0e30f;
  for (int i = tid; i < L; i += 128) mx = fmaxf(mx, r[i]);
  red[tid] = mx; __syncthreads();
  for (int off = 64; off > 0; off >>= 1) { if (tid < off) red[tid] = fmaxf(red[tid], red[tid + off]); __syncthreads(); }
  mx = red[0]; __syncthreads();
  float sum = 0.f;
  for (int i = tid; i < L; i += 128) { const float e = __expf(r[i] - mx); r[i] = e; sum += e; }
  red[tid] = sum; __syncthreads();
  for (int off = 64; off > 0; off >>= 1) { if (tid < off) red[tid] += red[tid + off]; __syncthreads(); }
  const float inv = 1.0f / red[0];
  for (int i = tid; i < L; i += 128) r[i] *= inv;
}

// ---------------------------------------------------------------------------
// Token embedding gather + positional add: out[b,s,:] = emb[tok[b,s]] + pos[s]
// ---------------------------------------------------------------------------
__global__ __launch_bounds__(128) void embed_kernel(
    const int* __restrict__ tok, const float* __restrict__ emb,
    const float* __restrict__ pos, float* __restrict__ out, int Sdim, int D) {
  const long long rs = blockIdx.x;       // b*S + s
  const int s = (int)(rs % Sdim);
  const long long t = tok[rs];
  const float* e  = emb + t * D;
  const float* pp = pos + (long long)s * D;
  float* o = out + rs * (long long)D;
  for (int d = threadIdx.x; d < D; d += 128) o[d] = e[d] + pp[d];
}

// ---------------------------------------------------------------------------
// Host-side helpers
// ---------------------------------------------------------------------------
static void launch_gemm(hipStream_t st, const float* A, const float* Bm, float* C,
                        const float* bias, const float* resid, const float* mask,
                        int M, int N, int K, int lda, int ldb, int ldc,
                        int Z, int zdiv,
                        long long sAh, long long sAl, long long sBh, long long sBl,
                        long long sCh, long long sCl, long long sbh, long long sbl,
                        long long sml, float scale, int causal, int epi, bool transB) {
  GemmP p;
  p.A = A; p.Bm = Bm; p.C = C; p.bias = bias; p.resid = resid; p.mask = mask;
  p.sA_hi = sAh; p.sA_lo = sAl; p.sB_hi = sBh; p.sB_lo = sBl;
  p.sC_hi = sCh; p.sC_lo = sCl; p.sBias_hi = sbh; p.sBias_lo = sbl; p.sMask_lo = sml;
  p.lda = lda; p.ldb = ldb; p.ldc = ldc; p.K = K; p.zdiv = zdiv;
  p.scale = scale; p.causal = causal;
  const dim3 block(256);
  if (N % 128 == 0) {
    const dim3 grid(N / 128, M / BM, Z);
    if (transB) {
      if (epi == EPI_SCORE) gemm_wmma_kernel<128, 2, 4, true,  EPI_SCORE><<<grid, block, 0, st>>>(p);
      else                  gemm_wmma_kernel<128, 2, 4, true,  EPI_BIAS ><<<grid, block, 0, st>>>(p);
    } else {
      if (epi == EPI_BIAS_RELU)       gemm_wmma_kernel<128, 2, 4, false, EPI_BIAS_RELU ><<<grid, block, 0, st>>>(p);
      else if (epi == EPI_BIAS_RESID) gemm_wmma_kernel<128, 2, 4, false, EPI_BIAS_RESID><<<grid, block, 0, st>>>(p);
      else if (epi == EPI_BIAS)       gemm_wmma_kernel<128, 2, 4, false, EPI_BIAS      ><<<grid, block, 0, st>>>(p);
      else                            gemm_wmma_kernel<128, 2, 4, false, EPI_NONE      ><<<grid, block, 0, st>>>(p);
    }
  } else {
    const dim3 grid(N / 64, M / BM, Z);
    if (epi == EPI_BIAS) gemm_wmma_kernel<64, 4, 2, false, EPI_BIAS><<<grid, block, 0, st>>>(p);
    else                 gemm_wmma_kernel<64, 4, 2, false, EPI_NONE><<<grid, block, 0, st>>>(p);
  }
}

struct AttnW { const float *wq, *bq, *wk, *bk, *wv, *bv, *wo, *bo; };

// Full multi-head attention: xout = resid + MHA(xq -> Q ; xkv -> K,V)
static void mha(hipStream_t st, const float* xq, const float* xkv, const AttnW& w,
                const float* mask, int causal, const float* resid, float* xout,
                float* Q, float* Kb, float* Vb, float* Sc, float* O) {
  const long long MS = (long long)B_ * S_;
  // Q/K/V projections, batched over heads (weights [H, D, DH] -> per-head [D,DH])
  launch_gemm(st, xq,  w.wq, Q,  w.bq, nullptr, nullptr, (int)MS, DH_, D_, D_, DH_, DH_,
              H_, 1, 0, 0, (long long)D_ * DH_, 0, MS * DH_, 0, DH_, 0, 0, 1.0f, 0, EPI_BIAS, false);
  launch_gemm(st, xkv, w.wk, Kb, w.bk, nullptr, nullptr, (int)MS, DH_, D_, D_, DH_, DH_,
              H_, 1, 0, 0, (long long)D_ * DH_, 0, MS * DH_, 0, DH_, 0, 0, 1.0f, 0, EPI_BIAS, false);
  launch_gemm(st, xkv, w.wv, Vb, w.bv, nullptr, nullptr, (int)MS, DH_, D_, D_, DH_, DH_,
              H_, 1, 0, 0, (long long)D_ * DH_, 0, MS * DH_, 0, DH_, 0, 0, 1.0f, 0, EPI_BIAS, false);
  // scores[h,b] = 1/sqrt(DH) * Q[h,b] @ K[h,b]^T  (+causal/padding masks)
  launch_gemm(st, Q, Kb, Sc, nullptr, nullptr, mask, S_, S_, DH_, DH_, DH_, S_,
              H_ * B_, B_, MS * DH_, (long long)S_ * DH_, MS * DH_, (long long)S_ * DH_,
              (long long)B_ * S_ * S_, (long long)S_ * S_, 0, 0, (long long)S_,
              0.125f, causal, EPI_SCORE, true);
  softmax_kernel<<<dim3(H_ * B_ * S_), dim3(128), 0, st>>>(Sc, S_);
  // O[b,s,h*DH+e] = probs[h,b] @ V[h,b]
  launch_gemm(st, Sc, Vb, O, nullptr, nullptr, nullptr, S_, DH_, S_, S_, DH_, H_ * DH_,
              H_ * B_, B_, (long long)B_ * S_ * S_, (long long)S_ * S_,
              MS * DH_, (long long)S_ * DH_, (long long)DH_, (long long)S_ * H_ * DH_,
              0, 0, 0, 1.0f, 0, EPI_NONE, false);
  // xout = resid + O @ wo + bo
  launch_gemm(st, O, w.wo, xout, w.bo, resid, nullptr, (int)MS, D_, H_ * DH_,
              H_ * DH_, D_, D_, 1, 1, 0, 0, 0, 0, 0, 0, 0, 0, 0, 1.0f, 0, EPI_BIAS_RESID, false);
}

// ---------------------------------------------------------------------------
// Input flattening map.
// Top level: setup_inputs() insertion order. Nested pytrees (lists of dicts)
// flattened jax-style: lists in order, dict keys sorted alphabetically.
// Encoder block leaves (16): attn{bk,bo,bq,bv,wk,wo,wq,wv}, b1,b2,
//   ln1b,ln1g,ln2b,ln2g, w1,w2.
// Decoder block leaves (26): b1,b2, cattn{8}, ln1b,ln1g,ln2b,ln2g,ln3b,ln3g,
//   sattn{8}, w1,w2.
// ---------------------------------------------------------------------------
extern "C" void kernel_launch(void* const* d_in, const int* in_sizes, int n_in,
                              void* d_out, int out_size, void* d_ws, size_t ws_size,
                              hipStream_t stream) {
  (void)in_sizes; (void)n_in; (void)out_size; (void)ws_size;

  const int*   src_tok  = (const int*)d_in[0];
  const int*   tgt_tok  = (const int*)d_in[1];
  /* d_in[2] = src_attention_mask: ignored by reference forward */
  const float* tgt_mask = (const float*)d_in[3];
  const float* emb      = (const float*)d_in[4];
  const float* enc_pos  = (const float*)d_in[5];
  const float* dec_pos  = (const float*)d_in[6];
  const int ENC0 = 7;                      // 16 leaves per encoder block
  const int DEC0 = ENC0 + 16 * NE_;        // 26 leaves per decoder block
  const int TAIL = DEC0 + 26 * ND_;
  const float* enc_ln_b = (const float*)d_in[TAIL + 0];
  const float* enc_ln_g = (const float*)d_in[TAIL + 1];
  const float* dec_ln_b = (const float*)d_in[TAIL + 2];
  const float* dec_ln_g = (const float*)d_in[TAIL + 3];
  const float* head_b   = (const float*)d_in[TAIL + 4];
  float* logits = (float*)d_out;

  // ---- workspace carve-out (~167 MB of f32) ----
  float* w = (float*)d_ws;
  size_t off = 0;
  auto carve = [&](size_t n) { float* p = w + off; off += n; return p; };
  const size_t MSD = (size_t)B_ * S_ * D_;
  float* xbuf   = carve(MSD);                         // encoder residual stream
  float* ybuf   = carve(MSD);                         // decoder residual stream
  float* xn     = carve(MSD);                         // LN output
  float* encout = carve(MSD);                         // final encoder output
  float* Q      = carve((size_t)H_ * B_ * S_ * DH_);
  float* Kb     = carve((size_t)H_ * B_ * S_ * DH_);
  float* Vb     = carve((size_t)H_ * B_ * S_ * DH_);
  float* O      = carve(MSD);                         // concat-head attention out
  float* ffb    = carve((size_t)B_ * S_ * F_);        // FFN hidden
  float* Sc     = carve((size_t)H_ * B_ * S_ * S_);   // attention scores/probs

  const long long MS = (long long)B_ * S_;
  const dim3 lnGrid((unsigned)MS), lnBlk(128);

  // ================= encoder =================
  embed_kernel<<<dim3((unsigned)MS), dim3(128), 0, stream>>>(src_tok, emb, enc_pos, xbuf, S_, D_);
  for (int i = 0; i < NE_; ++i) {
    const int e = ENC0 + 16 * i;
    AttnW aw{ (const float*)d_in[e + 6], (const float*)d_in[e + 2],   // wq, bq
              (const float*)d_in[e + 4], (const float*)d_in[e + 0],   // wk, bk
              (const float*)d_in[e + 7], (const float*)d_in[e + 3],   // wv, bv
              (const float*)d_in[e + 5], (const float*)d_in[e + 1] }; // wo, bo
    const float* ln1g = (const float*)d_in[e + 11]; const float* ln1b = (const float*)d_in[e + 10];
    const float* ln2g = (const float*)d_in[e + 13]; const float* ln2b = (const float*)d_in[e + 12];
    const float* w1 = (const float*)d_in[e + 14];   const float* b1 = (const float*)d_in[e + 8];
    const float* w2 = (const float*)d_in[e + 15];   const float* b2 = (const float*)d_in[e + 9];

    layernorm_kernel<<<lnGrid, lnBlk, 0, stream>>>(xbuf, ln1g, ln1b, xn, D_);
    mha(stream, xn, xn, aw, nullptr, 0, xbuf, xbuf, Q, Kb, Vb, Sc, O);
    layernorm_kernel<<<lnGrid, lnBlk, 0, stream>>>(xbuf, ln2g, ln2b, xn, D_);
    launch_gemm(stream, xn, w1, ffb, b1, nullptr, nullptr, (int)MS, F_, D_, D_, F_, F_,
                1, 1, 0, 0, 0, 0, 0, 0, 0, 0, 0, 1.0f, 0, EPI_BIAS_RELU, false);
    launch_gemm(stream, ffb, w2, xbuf, b2, xbuf, nullptr, (int)MS, D_, F_, F_, D_, D_,
                1, 1, 0, 0, 0, 0, 0, 0, 0, 0, 0, 1.0f, 0, EPI_BIAS_RESID, false);
  }
  layernorm_kernel<<<lnGrid, lnBlk, 0, stream>>>(xbuf, enc_ln_g, enc_ln_b, encout, D_);

  // ================= decoder =================
  embed_kernel<<<dim3((unsigned)MS), dim3(128), 0, stream>>>(tgt_tok, emb, dec_pos, ybuf, S_, D_);
  for (int i = 0; i < ND_; ++i) {
    const int d = DEC0 + 26 * i;
    AttnW sw{ (const float*)d_in[d + 22], (const float*)d_in[d + 18],   // sattn wq,bq
              (const float*)d_in[d + 20], (const float*)d_in[d + 16],   // wk,bk
              (const float*)d_in[d + 23], (const float*)d_in[d + 19],   // wv,bv
              (const float*)d_in[d + 21], (const float*)d_in[d + 17] }; // wo,bo
    AttnW cw{ (const float*)d_in[d + 8],  (const float*)d_in[d + 4],    // cattn wq,bq
              (const float*)d_in[d + 6],  (const float*)d_in[d + 2],    // wk,bk
              (const float*)d_in[d + 9],  (const float*)d_in[d + 5],    // wv,bv
              (const float*)d_in[d + 7],  (const float*)d_in[d + 3] };  // wo,bo
    const float* ln1g = (const float*)d_in[d + 11]; const float* ln1b = (const float*)d_in[d + 10];
    const float* ln2g = (const float*)d_in[d + 13]; const float* ln2b = (const float*)d_in[d + 12];
    const float* ln3g = (const float*)d_in[d + 15]; const float* ln3b = (const float*)d_in[d + 14];
    const float* w1 = (const float*)d_in[d + 24];   const float* b1 = (const float*)d_in[d + 0];
    const float* w2 = (const float*)d_in[d + 25];   const float* b2 = (const float*)d_in[d + 1];

    layernorm_kernel<<<lnGrid, lnBlk, 0, stream>>>(ybuf, ln1g, ln1b, xn, D_);
    mha(stream, xn, xn, sw, tgt_mask, /*causal=*/1, ybuf, ybuf, Q, Kb, Vb, Sc, O);
    layernorm_kernel<<<lnGrid, lnBlk, 0, stream>>>(ybuf, ln2g, ln2b, xn, D_);
    mha(stream, xn, encout, cw, tgt_mask, /*causal=*/0, ybuf, ybuf, Q, Kb, Vb, Sc, O);
    layernorm_kernel<<<lnGrid, lnBlk, 0, stream>>>(ybuf, ln3g, ln3b, xn, D_);
    launch_gemm(stream, xn, w1, ffb, b1, nullptr, nullptr, (int)MS, F_, D_, D_, F_, F_,
                1, 1, 0, 0, 0, 0, 0, 0, 0, 0, 0, 1.0f, 0, EPI_BIAS_RELU, false);
    launch_gemm(stream, ffb, w2, ybuf, b2, ybuf, nullptr, (int)MS, D_, F_, F_, D_, D_,
                1, 1, 0, 0, 0, 0, 0, 0, 0, 0, 0, 1.0f, 0, EPI_BIAS_RESID, false);
  }
  layernorm_kernel<<<lnGrid, lnBlk, 0, stream>>>(ybuf, dec_ln_g, dec_ln_b, xn, D_);

  // ============ tied LM head: logits = y @ emb^T + head_b ============
  launch_gemm(stream, xn, emb, logits, head_b, nullptr, nullptr, (int)MS, V_, D_,
              D_, D_, V_, 1, 1, 0, 0, 0, 0, 0, 0, 0, 0, 0, 1.0f, 0, EPI_BIAS, /*transB=*/true);
}
